// MultiheadAttention_197568495758
// MI455X (gfx1250) — matrix operand order repaired and checked by reading, not compiled, not run on hardware
//
#include <hip/hip_runtime.h>

// ---------------------------------------------------------------------------
// Flash-attention forward for MI455X (gfx1250, wave32, WMMA).
// B=4, H=16, S=2048, D=64, fp32 in/out; f16 WMMA, f32 accumulate.
// KVT=64 double-buffered tiles, DPP row-max, row-sum via ones-matrix WMMA,
// log2-domain softmax (1/sqrt(D)*log2e folded into Q).
// ---------------------------------------------------------------------------

typedef __attribute__((ext_vector_type(16))) _Float16 v16h;
typedef __attribute__((ext_vector_type(8)))  _Float16 h8;
typedef __attribute__((ext_vector_type(2)))  _Float16 h2;
typedef __attribute__((ext_vector_type(8)))  float    v8f;
typedef __attribute__((ext_vector_type(4)))  float    f4;

#define BB   4
#define SS   2048
#define EE   1024
#define HH   16
#define DD   64
#define KVT  64                       // KV positions per tile
#define NIT  (SS / KVT)               // 32 iterations
#define LDK  72                       // LDS row stride (halves): 144B, 16B-aligned
// 1/sqrt(D) * log2(e) folded into Q at f16 conversion -> scores in log2 domain
#define QSCALE (0.125f * 1.44269504088896f)

// Butterfly max over the 16 lanes of a half-wave using DPP (pure VALU).
// Generators xor1, xor2, xor7(row_half_mirror), xor15(row_mirror) cover all 16.
static __device__ __forceinline__ float dpp_fmax16(float x) {
  float t;
  t = __int_as_float(__builtin_amdgcn_update_dpp(0, __float_as_int(x), 0xB1,  0xF, 0xF, true)); x = fmaxf(x, t);
  t = __int_as_float(__builtin_amdgcn_update_dpp(0, __float_as_int(x), 0x4E,  0xF, 0xF, true)); x = fmaxf(x, t);
  t = __int_as_float(__builtin_amdgcn_update_dpp(0, __float_as_int(x), 0x141, 0xF, 0xF, true)); x = fmaxf(x, t);
  t = __int_as_float(__builtin_amdgcn_update_dpp(0, __float_as_int(x), 0x140, 0xF, 0xF, true)); x = fmaxf(x, t);
  return x;
}

__global__ __launch_bounds__(128)
void mha_flash_fwd(const float* __restrict__ Q, const float* __restrict__ K,
                   const float* __restrict__ V, float* __restrict__ O) {
  __shared__ alignas(16) _Float16 sK [2][KVT * LDK];   // K tile  [kv][d], f16
  __shared__ alignas(16) _Float16 sVT[2][DD  * LDK];   // V tile transposed [d][kv]
  __shared__ alignas(16) _Float16 sP [4][16 * LDK];    // per-wave P relayout scratch

  const int tid  = threadIdx.x;
  const int wave = tid >> 5;
  const int lane = tid & 31;
  const int half = lane >> 4;       // 0 -> lanes 0-15, 1 -> lanes 16-31
  const int l16  = lane & 15;

  // block -> (batch, head, 64-row q tile); each wave owns 16 q rows.
  const int blocksPerSeq = SS / 64;                  // 32
  const int b   = blockIdx.x / (HH * blocksPerSeq);
  const int rem = blockIdx.x % (HH * blocksPerSeq);
  const int h   = rem / blocksPerSeq;
  const int q0  = (rem % blocksPerSeq) * 64 + wave * 16;

  const float* Qbase = Q + ((size_t)(b * SS + q0) * EE + h * DD);
  const float* Kbase = K + ((size_t)b * SS * EE + h * DD);
  const float* Vbase = V + ((size_t)b * SS * EE + h * DD);

  // Constant all-ones B fragment: P x Ones(32x16) puts rowsum(P) in every
  // column of the C fragment => per-lane VGPR r holds the (broadcast) row sum.
  v16h ones;
#pragma unroll
  for (int j = 0; j < 16; ++j) ones[j] = (_Float16)1.0f;

  // ---- Q A-fragments (16x32 f16 A-layout; two K=32 chunks over D=64) ------
  // A-layout (ISA 7.12.2): lanes<16 row=l16, elems 0..7 = K lo..lo+7,
  // elems 8..15 = K lo+16..+23; lanes>=16 same row, K offsets +8.
  v16h qa[2];
  {
    const float* qrow = Qbase + (size_t)l16 * EE;
#pragma unroll
    for (int c = 0; c < 2; ++c) {
      const int lo = c * 32 + half * 8;
      const int hi = c * 32 + 16 + half * 8;
      f4 a0 = *(const f4*)(qrow + lo);
      f4 a1 = *(const f4*)(qrow + lo + 4);
      f4 b0 = *(const f4*)(qrow + hi);
      f4 b1 = *(const f4*)(qrow + hi + 4);
      v16h f;
#pragma unroll
      for (int j = 0; j < 4; ++j) {
        f[j]      = (_Float16)(a0[j] * QSCALE);
        f[4 + j]  = (_Float16)(a1[j] * QSCALE);
        f[8 + j]  = (_Float16)(b0[j] * QSCALE);
        f[12 + j] = (_Float16)(b1[j] * QSCALE);
      }
      qa[c] = f;
    }
  }

  v8f acc_o[4];
  v8f l_v = {0.f, 0.f, 0.f, 0.f, 0.f, 0.f, 0.f, 0.f};
  float m_i[8];
#pragma unroll
  for (int r = 0; r < 8; ++r) m_i[r] = -1e30f;
#pragma unroll
  for (int t = 0; t < 4; ++t) {
    v8f z = {0.f, 0.f, 0.f, 0.f, 0.f, 0.f, 0.f, 0.f};
    acc_o[t] = z;
  }

  // Cooperative tile staging: thread owns kv rows {2p, 2p+1}, d cols col0..+15.
  const int p    = tid & 31;
  const int col0 = (tid >> 5) * 16;
  const int kvr  = 2 * p;

  f4 kst[8], vst[8];   // staged fp32 tile (loads issued early, cvt deferred)

  // -------- prologue: stage tile 0 -----------------------------------------
  {
    const float* kr = Kbase + (size_t)kvr * EE + col0;
    const float* vr = Vbase + (size_t)kvr * EE + col0;
#pragma unroll
    for (int i = 0; i < 4; ++i) {
      kst[i]     = ((const f4*)kr)[i];
      kst[4 + i] = ((const f4*)(kr + EE))[i];
      vst[i]     = ((const f4*)vr)[i];
      vst[4 + i] = ((const f4*)(vr + EE))[i];
    }
  }

  for (int it = 0; it < NIT; ++it) {
    const int bb = it & 1;

    // ---- store staged tile into LDS buffer bb (cvt f32->f16) -------------
    {
      h8 a0, a1, c0, c1;
#pragma unroll
      for (int j = 0; j < 4; ++j) {
        a0[j] = (_Float16)kst[0][j]; a0[4 + j] = (_Float16)kst[1][j];
        a1[j] = (_Float16)kst[2][j]; a1[4 + j] = (_Float16)kst[3][j];
        c0[j] = (_Float16)kst[4][j]; c0[4 + j] = (_Float16)kst[5][j];
        c1[j] = (_Float16)kst[6][j]; c1[4 + j] = (_Float16)kst[7][j];
      }
      *(h8*)(&sK[bb][kvr * LDK + col0])           = a0;
      *(h8*)(&sK[bb][kvr * LDK + col0 + 8])       = a1;
      *(h8*)(&sK[bb][(kvr + 1) * LDK + col0])     = c0;
      *(h8*)(&sK[bb][(kvr + 1) * LDK + col0 + 8]) = c1;
#pragma unroll
      for (int j = 0; j < 16; ++j) {              // transpose V, packed half2
        h2 pk;
        pk[0] = (_Float16)vst[j >> 2][j & 3];       // row kvr
        pk[1] = (_Float16)vst[4 + (j >> 2)][j & 3]; // row kvr+1
        *(h2*)(&sVT[bb][(col0 + j) * LDK + kvr]) = pk;  // ds_store_b32
      }
    }

    // ---- issue global loads for next tile (wait deferred past compute) ----
    if (it + 1 < NIT) {
      const float* kr = Kbase + (size_t)((it + 1) * KVT + kvr) * EE + col0;
      const float* vr = Vbase + (size_t)((it + 1) * KVT + kvr) * EE + col0;
#pragma unroll
      for (int i = 0; i < 4; ++i) {
        kst[i]     = ((const f4*)kr)[i];
        kst[4 + i] = ((const f4*)(kr + EE))[i];
        vst[i]     = ((const f4*)vr)[i];
        vst[4 + i] = ((const f4*)(vr + EE))[i];
      }
    }

    __syncthreads();   // tile bb visible to all waves; prev readers done

    // ---- S = Q*K^T : 16x64 scores, four 16x16 C-frags, 8 WMMAs ------------
    // B-layout (matches A table order): lane n=l16; elems 0..7 = K half*8..+7,
    // elems 8..15 = K 16+half*8..+7.
    v8f accs[4];
#pragma unroll
    for (int nt = 0; nt < 4; ++nt) {
      v8f acc = {0.f, 0.f, 0.f, 0.f, 0.f, 0.f, 0.f, 0.f};
#pragma unroll
      for (int c = 0; c < 2; ++c) {
        const int base = (nt * 16 + l16) * LDK + c * 32 + half * 8;
        h8 blo = *(const h8*)(&sK[bb][base]);
        h8 bhi = *(const h8*)(&sK[bb][base + 16]);
        v16h kb = __builtin_shufflevector(blo, bhi, 0, 1, 2, 3, 4, 5, 6, 7,
                                          8, 9, 10, 11, 12, 13, 14, 15);
        acc = __builtin_amdgcn_wmma_f32_16x16x32_f16(
            false, qa[c], false, kb, (short)0, acc, false, false);
      }
      accs[nt] = acc;
    }

    // ---- online softmax: row max via DPP, exp2, stage P in A-layout -------
    v8f alpha_v;
#pragma unroll
    for (int r = 0; r < 8; ++r) {
      const float s0 = accs[0][r], s1 = accs[1][r];
      const float s2 = accs[2][r], s3 = accs[3][r];
      float mx = fmaxf(fmaxf(s0, s1), fmaxf(s2, s3));
      mx = dpp_fmax16(mx);                       // row max across 16 lanes
      const float mn = fmaxf(m_i[r], mx);
      alpha_v[r] = __builtin_amdgcn_exp2f(m_i[r] - mn);
      m_i[r]     = mn;
      const float p0 = __builtin_amdgcn_exp2f(s0 - mn);
      const float p1 = __builtin_amdgcn_exp2f(s1 - mn);
      const float p2 = __builtin_amdgcn_exp2f(s2 - mn);
      const float p3 = __builtin_amdgcn_exp2f(s3 - mn);
      const int mrow = r + half * 8;             // C-layout row of VGPR r
      sP[wave][mrow * LDK + l16]      = (_Float16)p0;   // relayout C -> A
      sP[wave][mrow * LDK + 16 + l16] = (_Float16)p1;
      sP[wave][mrow * LDK + 32 + l16] = (_Float16)p2;
      sP[wave][mrow * LDK + 48 + l16] = (_Float16)p3;
    }
    __builtin_amdgcn_wave_barrier();  // DS in-order per wave; pin sched order

    v16h pa[2];                        // P as A-matrix (16x64, two K=32 frags)
#pragma unroll
    for (int c = 0; c < 2; ++c) {
      const int base = l16 * LDK + c * 32 + half * 8;
      h8 plo = *(const h8*)(&sP[wave][base]);
      h8 phi = *(const h8*)(&sP[wave][base + 16]);
      pa[c] = __builtin_shufflevector(plo, phi, 0, 1, 2, 3, 4, 5, 6, 7,
                                      8, 9, 10, 11, 12, 13, 14, 15);
    }

    // ---- row sums on the matrix core: SUM = P x Ones (2 WMMAs) ------------
    // Every column of Ones is 1 => C[m][n] = rowsum(P[m]) for all n, i.e. the
    // sum arrives broadcast across lanes in the exact l_v layout.
    {
      v8f sum = {0.f, 0.f, 0.f, 0.f, 0.f, 0.f, 0.f, 0.f};
      sum = __builtin_amdgcn_wmma_f32_16x16x32_f16(
          false, pa[0], false, ones, (short)0, sum, false, false);
      sum = __builtin_amdgcn_wmma_f32_16x16x32_f16(
          false, pa[1], false, ones, (short)0, sum, false, false);
      l_v = l_v * alpha_v + sum;
    }

    // ---- O = O*alpha + P*V : four 16x16 d-tiles x two K=32 chunks, 8 WMMAs -
#pragma unroll
    for (int t = 0; t < 4; ++t) {
      v8f ot = acc_o[t] * alpha_v;
#pragma unroll
      for (int c = 0; c < 2; ++c) {
        const int base = (t * 16 + l16) * LDK + c * 32 + half * 8;
        h8 vlo = *(const h8*)(&sVT[bb][base]);
        h8 vhi = *(const h8*)(&sVT[bb][base + 16]);
        v16h vb = __builtin_shufflevector(vlo, vhi, 0, 1, 2, 3, 4, 5, 6, 7,
                                          8, 9, 10, 11, 12, 13, 14, 15);
        ot = __builtin_amdgcn_wmma_f32_16x16x32_f16(
            false, pa[c], false, vb, (short)0, ot, false, false);
      }
      acc_o[t] = ot;
    }
  }

  // ---- epilogue: O / l, write fp32 ----------------------------------------
  float* Obase = O + ((size_t)(b * SS + q0) * EE + h * DD);
#pragma unroll
  for (int r = 0; r < 8; ++r) {
    const float inv  = 1.0f / l_v[r];
    const int   mrow = r + half * 8;
    float* orow = Obase + (size_t)mrow * EE;
#pragma unroll
    for (int t = 0; t < 4; ++t)
      orow[t * 16 + l16] = acc_o[t][r] * inv;
  }
}

extern "C" void kernel_launch(void* const* d_in, const int* in_sizes, int n_in,
                              void* d_out, int out_size, void* d_ws, size_t ws_size,
                              hipStream_t stream) {
  (void)in_sizes; (void)n_in; (void)d_ws; (void)ws_size; (void)out_size;
  const float* q = (const float*)d_in[0];
  const float* k = (const float*)d_in[1];
  const float* v = (const float*)d_in[2];
  float* out = (float*)d_out;
  const int grid = BB * HH * (SS / 64);   // 2048 blocks of 4 waves
  mha_flash_fwd<<<grid, 128, 0, stream>>>(q, k, v, out);
}
